// VanillaMHA_5085241278732
// MI455X (gfx1250) — matrix operand order repaired and checked by reading, not compiled, not run on hardware
//
#include <hip/hip_runtime.h>
#include <hip/hip_bf16.h>
#include <math.h>

// Problem constants (match reference)
#define BQ  4
#define TQ  2048
#define DQ  1024
#define HQ  16
#define DHQ 64

typedef __attribute__((ext_vector_type(16))) __bf16        bf16x16;
typedef __attribute__((ext_vector_type(8)))  float         f32x8;
typedef __attribute__((ext_vector_type(4)))  unsigned int  u32x4;

union FragBF {
  bf16x16 v;
  u32x4   u[2];   // 32 bytes
};

__device__ __forceinline__ unsigned short f2bf(float f) {
  unsigned int u = __float_as_uint(f);
  u += 0x7FFFu + ((u >> 16) & 1u);          // round-to-nearest-even
  return (unsigned short)(u >> 16);
}

__device__ __forceinline__ f32x8 wmma_bf16(const FragBF& a, const FragBF& b, f32x8 c) {
  return __builtin_amdgcn_wmma_f32_16x16x32_bf16(false, a.v, false, b.v, (short)0, c,
                                                 false, false);
}

// CDNA5 async global->LDS copy (16B per lane), tracked by ASYNCcnt.
__device__ __forceinline__ void async_copy_b128(void* lds_ptr, const void* gptr) {
  unsigned loff = (unsigned)(unsigned long long)lds_ptr;   // LDS byte offset
  asm volatile("global_load_async_to_lds_b128 %0, %1, off"
               :: "v"(loff), "v"(gptr));
}
__device__ __forceinline__ void wait_async0() {
  asm volatile("s_wait_asynccnt 0x0" ::: "memory");
}

// xor-butterfly within each 16-lane half via v_permlane16_b32 (pure VALU,
// co-executes with XDL WMMA); fallback to shfl_xor (ds_bpermute) if absent.
constexpr unsigned make_sel(int base, int xm) {
  unsigned s = 0;
  for (int i = 0; i < 8; ++i) s |= (unsigned)(((base + i) ^ xm) & 15) << (4 * i);
  return s;
}
template <int XM>
__device__ __forceinline__ float xor16(float v) {
#if __has_builtin(__builtin_amdgcn_permlane16)
  constexpr unsigned S0 = make_sel(0, XM);
  constexpr unsigned S1 = make_sel(8, XM);
  int i = __builtin_amdgcn_permlane16(__float_as_int(v), __float_as_int(v),
                                      (int)S0, (int)S1, false, false);
  return __int_as_float(i);
#else
  return __shfl_xor(v, XM, 32);
#endif
}

// ---------------------------------------------------------------------------
// Kernel 1: fp32 weight (N x K, row-major) -> bf16 transposed weight (K x N)
// ---------------------------------------------------------------------------
__global__ void wt_kernel(const float* __restrict__ W, unsigned short* __restrict__ Wt,
                          int K, int N) {
  __shared__ unsigned short tile[32][33];
  const int nTilesN = N >> 5;
  const int n0 = (blockIdx.x % nTilesN) << 5;
  const int k0 = (blockIdx.x / nTilesN) << 5;
  const int t = threadIdx.x;
  {
    int r = t >> 3, seg = (t & 7) << 2;
    float4 w = *reinterpret_cast<const float4*>(W + (size_t)(n0 + r) * K + k0 + seg);
    tile[r][seg + 0] = f2bf(w.x);
    tile[r][seg + 1] = f2bf(w.y);
    tile[r][seg + 2] = f2bf(w.z);
    tile[r][seg + 3] = f2bf(w.w);
  }
  __syncthreads();
  {
    int kk = t >> 3, ns = (t & 7) << 2;
    uint2 p;
    p.x = (unsigned)tile[ns + 0][kk] | ((unsigned)tile[ns + 1][kk] << 16);
    p.y = (unsigned)tile[ns + 2][kk] | ((unsigned)tile[ns + 3][kk] << 16);
    *reinterpret_cast<uint2*>(Wt + (size_t)(k0 + kk) * N + n0 + ns) = p;
  }
}

// ---------------------------------------------------------------------------
// Kernel 2: fused LayerNorm (fp32 math) -> bf16 activations
// ---------------------------------------------------------------------------
__global__ void ln_kernel(const float* __restrict__ x, const float* __restrict__ g,
                          const float* __restrict__ be, unsigned short* __restrict__ xn) {
  const int row = blockIdx.x;
  const int t = threadIdx.x;
  const float4 xv = *reinterpret_cast<const float4*>(x + (size_t)row * DQ + t * 4);
  float s  = xv.x + xv.y + xv.z + xv.w;
  float s2 = xv.x * xv.x + xv.y * xv.y + xv.z * xv.z + xv.w * xv.w;
  #pragma unroll
  for (int off = 16; off; off >>= 1) {
    s  += __shfl_down(s,  off, 32);
    s2 += __shfl_down(s2, off, 32);
  }
  __shared__ float ws1[8], ws2[8];
  const int lane = t & 31, w = t >> 5;
  if (lane == 0) { ws1[w] = s; ws2[w] = s2; }
  __syncthreads();
  if (t == 0) {
    float a = 0.f, b2 = 0.f;
    #pragma unroll
    for (int i = 0; i < 8; ++i) { a += ws1[i]; b2 += ws2[i]; }
    ws1[0] = a; ws2[0] = b2;
  }
  __syncthreads();
  const float mean = ws1[0] * (1.0f / DQ);
  const float var  = ws2[0] * (1.0f / DQ) - mean * mean;
  const float rstd = rsqrtf(var + 1e-5f);
  const float4 gv = *reinterpret_cast<const float4*>(g  + t * 4);
  const float4 bv = *reinterpret_cast<const float4*>(be + t * 4);
  const float y0 = (xv.x - mean) * rstd * gv.x + bv.x;
  const float y1 = (xv.y - mean) * rstd * gv.y + bv.y;
  const float y2 = (xv.z - mean) * rstd * gv.z + bv.z;
  const float y3 = (xv.w - mean) * rstd * gv.w + bv.w;
  uint2 p;
  p.x = (unsigned)f2bf(y0) | ((unsigned)f2bf(y1) << 16);
  p.y = (unsigned)f2bf(y2) | ((unsigned)f2bf(y3) << 16);
  *reinterpret_cast<uint2*>(xn + (size_t)row * DQ + t * 4) = p;
}

// ---------------------------------------------------------------------------
// Kernel 3/5: bf16 GEMM, C = A(MxK) * Bt(KxN), fp32 accumulation.
// Block tile 128x256, K-step 32, 8 waves (2x4), wave tile 64x64:
// 16 x v_wmma_f32_16x16x32_bf16 per wave per K-step against 8 A b128 + 64 B u16 ds_loads.
// Double-buffered async global->LDS pipeline (ASYNCcnt), 1 barrier/step.
// MODE 0: scatter bf16 into qkv buffer (Q pre-scaled by 1/sqrt(DH),
//         K into transposed Kt); MODE 1: fp32 store.
// ---------------------------------------------------------------------------
template <int MODE>
__global__ void gemm_kernel(const unsigned short* __restrict__ A,
                            const unsigned short* __restrict__ Bt,
                            unsigned short* __restrict__ qkv,
                            unsigned short* __restrict__ Kt,
                            float* __restrict__ Cf,
                            int M, int N, int K) {
  __shared__ unsigned short lA[2][128 * 32];   // [m][k]  2 x  8 KB
  __shared__ unsigned short lB[2][32 * 256];   // [k][n]  2 x 16 KB
  const int tid  = threadIdx.x;
  const int lane = tid & 31, wave = tid >> 5;
  const int wm = wave >> 2, wn = wave & 3;
  const int n0 = blockIdx.x * 256, m0 = blockIdx.y * 128;

  // per-thread staging slots: 2 x 16B for A, 4 x 16B for B
  int rA[2], cA[2], rB[4], cB[4];
  const unsigned short* gA[2];
  const unsigned short* gB[4];
  #pragma unroll
  for (int c = 0; c < 2; ++c) {
    const int idx = tid * 2 + c;
    rA[c] = idx >> 2;  cA[c] = (idx & 3) << 3;
    gA[c] = A + (size_t)(m0 + rA[c]) * K + cA[c];
  }
  #pragma unroll
  for (int c = 0; c < 4; ++c) {
    const int idx = tid * 4 + c;
    rB[c] = idx >> 5;  cB[c] = (idx & 31) << 3;
    gB[c] = Bt + (size_t)rB[c] * N + n0 + cB[c];
  }
  // prologue: issue tile 0 into buffer 0
  #pragma unroll
  for (int c = 0; c < 2; ++c)
    async_copy_b128(&lA[0][rA[c] * 32 + cA[c]], gA[c]);
  #pragma unroll
  for (int c = 0; c < 4; ++c)
    async_copy_b128(&lB[0][rB[c] * 256 + cB[c]], gB[c]);

  f32x8 acc[4][4];
  #pragma unroll
  for (int i = 0; i < 4; ++i)
    #pragma unroll
    for (int j = 0; j < 4; ++j)
      acc[i][j] = (f32x8){0.f, 0.f, 0.f, 0.f, 0.f, 0.f, 0.f, 0.f};

  const int kTiles = K >> 5;
  const int khalf = (lane >> 4) << 3;  // A-fragment K-chunk select (ISA layout)
  for (int kt = 0; kt < kTiles; ++kt) {
    const int cur = kt & 1;
    wait_async0();       // my buffer-cur loads have landed in LDS
    __syncthreads();     // everyone's landed; prev reads of buf cur^1 done
    if (kt + 1 < kTiles) {
      const int nxt = cur ^ 1;
      #pragma unroll
      for (int c = 0; c < 2; ++c)
        async_copy_b128(&lA[nxt][rA[c] * 32 + cA[c]],
                        gA[c] + (size_t)(kt + 1) * 32);
      #pragma unroll
      for (int c = 0; c < 4; ++c)
        async_copy_b128(&lB[nxt][rB[c] * 256 + cB[c]],
                        gB[c] + (size_t)(kt + 1) * 32 * N);
    }
    FragBF bfr[4];
    #pragma unroll
    for (int nt = 0; nt < 4; ++nt) {   // lane%16 = N index, 16 K per lane (table order)
      #pragma unroll
      for (int s = 0; s < 16; ++s) {
        const int kk = (s & 7) + khalf + ((s >> 3) << 4);
        bfr[nt].v[s] = __builtin_bit_cast(__bf16, lB[cur][kk * 256 + wn * 64 + nt * 16 + (lane & 15)]);
      }
    }
    #pragma unroll
    for (int mt = 0; mt < 4; ++mt) {
      FragBF afr;
      const int row = wm * 64 + mt * 16 + (lane & 15);
      afr.u[0] = *reinterpret_cast<const u32x4*>(&lA[cur][row * 32 + khalf]);
      afr.u[1] = *reinterpret_cast<const u32x4*>(&lA[cur][row * 32 + 16 + khalf]);
      #pragma unroll
      for (int nt = 0; nt < 4; ++nt)
        acc[mt][nt] = wmma_bf16(afr, bfr[nt], acc[mt][nt]);
    }
  }

  const int hi = lane >> 4;
  #pragma unroll
  for (int mt = 0; mt < 4; ++mt)
    #pragma unroll
    for (int nt = 0; nt < 4; ++nt) {
      const int n = n0 + wn * 64 + nt * 16 + (lane & 15);
      const int h = n / 192, sub = n - h * 192;     // (H, 3*DH) view (MODE 0)
      #pragma unroll
      for (int r = 0; r < 8; ++r) {
        const int m = m0 + wm * 64 + mt * 16 + r + hi * 8;
        float v = acc[mt][nt][r];
        if (MODE == 1) {
          Cf[(size_t)m * N + n] = v;
        } else {
          if (sub < 64) v *= 0.125f;                // fold 1/sqrt(DH) into Q
          const unsigned short bv = f2bf(v);
          if (sub >= 64 && sub < 128) {             // K -> transposed (B,H,DH,T)
            const int b = m >> 11, t = m & (TQ - 1);
            Kt[(((size_t)b * HQ + h) * DHQ + (sub - 64)) * TQ + t] = bv;
          } else {                                  // Q / V in (B,T,H,192)
            qkv[((size_t)m * HQ + h) * 192 + sub] = bv;
          }
        }
      }
    }
}

// ---------------------------------------------------------------------------
// Kernel 4: causal flash attention. Block = (b, h, 128 q-rows), 8 waves,
// wave owns 16 q-rows. Q (pre-scaled) pinned in registers, K/V double-
// buffered async into LDS, online softmax in fp32 with permlane16 row
// reductions, P through per-wave LDS into the WMMA A layout.
// ---------------------------------------------------------------------------
__global__ void attn_kernel(const unsigned short* __restrict__ qkv,
                            const unsigned short* __restrict__ Kt,
                            unsigned short* __restrict__ ctx) {
  __shared__ unsigned short lK[2][64 * 32];     // [d][t]  2 x 4 KB
  __shared__ unsigned short lV[2][32 * 64];     // [j][d]  2 x 4 KB
  __shared__ unsigned short lP[8 * 16 * 32];    // per-wave P, 8 KB

  const int tid = threadIdx.x, lane = tid & 31, wave = tid >> 5;
  const int qblk = blockIdx.x;                  // 0..15
  const int bh = blockIdx.y;                    // 0..63
  const int b = bh >> 4, h = bh & 15;
  const int q0 = qblk * 128;
  const int qrow_lo = q0 + wave * 16;
  const int hi = lane >> 4;

  // staging slots (one 16B chunk per thread per buffer)
  const int krow = tid >> 2, kseg = (tid & 3) << 3;
  const int vrow = tid >> 3, vseg = (tid & 7) << 3;
  const unsigned short* gK = Kt + ((size_t)bh * DHQ + krow) * TQ + kseg;
  const unsigned short* gV =
      qkv + ((size_t)(b * TQ + vrow) * HQ + h) * 192 + 128 + vseg;

  // preload Q fragments (two K=32 chains over head dim 64); Q is pre-scaled
  FragBF qf[2];
  {
    const int q = qrow_lo + (lane & 15);
    const size_t base = ((size_t)(b * TQ + q) * HQ + h) * 192;
    const int khalf = (lane >> 4) << 3;
    #pragma unroll
    for (int c = 0; c < 2; ++c) {
      qf[c].u[0] = *reinterpret_cast<const u32x4*>(qkv + base + c * 32 + khalf);
      qf[c].u[1] = *reinterpret_cast<const u32x4*>(qkv + base + c * 32 + 16 + khalf);
    }
  }
  // pin Q fragments in VGPRs so the loads cannot be sunk into the j-loop
  asm volatile("" : "+v"(qf[0].u[0]), "+v"(qf[0].u[1]),
                    "+v"(qf[1].u[0]), "+v"(qf[1].u[1]));

  f32x8 o[4];
  #pragma unroll
  for (int i = 0; i < 4; ++i) o[i] = (f32x8){0.f, 0.f, 0.f, 0.f, 0.f, 0.f, 0.f, 0.f};
  float rmax[8], rsum[8];
  #pragma unroll
  for (int r = 0; r < 8; ++r) { rmax[r] = -1e30f; rsum[r] = 0.f; }

  const int nJ = (q0 + 128) >> 5;   // 32-key steps up to causal frontier

  // prologue: stage tile 0 into buffer 0
  async_copy_b128(&lK[0][krow * 32 + kseg], gK);
  async_copy_b128(&lV[0][vrow * 64 + vseg], gV);

  for (int jt = 0; jt < nJ; ++jt) {
    const int cur = jt & 1;
    const int j0 = jt << 5;
    wait_async0();
    __syncthreads();
    if (jt + 1 < nJ) {
      const int nxt = cur ^ 1;
      async_copy_b128(&lK[nxt][krow * 32 + kseg], gK + ((jt + 1) << 5));
      async_copy_b128(&lV[nxt][vrow * 64 + vseg],
                      gV + (size_t)((jt + 1) << 5) * HQ * 192);
    }
    if (j0 > qrow_lo + 15) continue;        // whole tile above diagonal: skip
    const bool needMask = (j0 + 31 > qrow_lo);

    // S = Q * K^T : two 16x16 n-tiles, K-chained over d
    f32x8 s0 = (f32x8){0.f, 0.f, 0.f, 0.f, 0.f, 0.f, 0.f, 0.f};
    f32x8 s1 = s0;
    #pragma unroll
    for (int c = 0; c < 2; ++c) {
      FragBF k0f, k1f;
      #pragma unroll
      for (int s = 0; s < 16; ++s) {  // lane%16 = t, elements over d (table order)
        const int kk = (s & 7) + (hi << 3) + ((s >> 3) << 4);
        const unsigned short* kb = &lK[cur][(c * 32 + kk) * 32 + (lane & 15)];
        k0f.v[s] = __builtin_bit_cast(__bf16, kb[0]);
        k1f.v[s] = __builtin_bit_cast(__bf16, kb[16]);
      }
      s0 = wmma_bf16(qf[c], k0f, s0);
      s1 = wmma_bf16(qf[c], k1f, s1);
    }

    // online softmax (fp32); row reductions across 16-lane half groups
    const int colbase = j0 + (lane & 15);
    #pragma unroll
    for (int r = 0; r < 8; ++r) {
      const int row = qrow_lo + r + hi * 8;
      float v0 = s0[r];
      float v1 = s1[r];
      if (needMask) {
        if (colbase > row)      v0 = -1e30f;
        if (colbase + 16 > row) v1 = -1e30f;
      }
      float m = fmaxf(v0, v1);
      m = fmaxf(m, xor16<1>(m));
      m = fmaxf(m, xor16<2>(m));
      m = fmaxf(m, xor16<4>(m));
      m = fmaxf(m, xor16<8>(m));
      const float mnew = fmaxf(rmax[r], m);
      const float fct  = __expf(rmax[r] - mnew);
      const float p0 = __expf(v0 - mnew);
      const float p1 = __expf(v1 - mnew);
      float ps = p0 + p1;
      ps += xor16<1>(ps);
      ps += xor16<2>(ps);
      ps += xor16<4>(ps);
      ps += xor16<8>(ps);
      rsum[r] = rsum[r] * fct + ps;
      rmax[r] = mnew;
      o[0][r] *= fct; o[1][r] *= fct; o[2][r] *= fct; o[3][r] *= fct;
      const int prow = r + hi * 8;      // C-layout -> A-layout via LDS
      lP[wave * 512 + prow * 32 + (lane & 15)]      = f2bf(p0);
      lP[wave * 512 + prow * 32 + 16 + (lane & 15)] = f2bf(p1);
    }
    asm volatile("s_wait_dscnt 0x0" ::: "memory");  // same-wave LDS RAW

    // ctx += P * V
    FragBF pf;
    {
      const unsigned short* pb = &lP[wave * 512 + (lane & 15) * 32];
      const int kh = hi << 3;
      pf.u[0] = *reinterpret_cast<const u32x4*>(pb + kh);
      pf.u[1] = *reinterpret_cast<const u32x4*>(pb + 16 + kh);
    }
    #pragma unroll
    for (int dt = 0; dt < 4; ++dt) {
      FragBF vf;
      #pragma unroll
      for (int s = 0; s < 16; ++s) {  // lane%16 = d, elements over j (table order)
        const int kk = (s & 7) + (hi << 3) + ((s >> 3) << 4);
        vf.v[s] = __builtin_bit_cast(__bf16, lV[cur][kk * 64 + dt * 16 + (lane & 15)]);
      }
      o[dt] = wmma_bf16(pf, vf, o[dt]);
    }
  }

  // finalize: divide by row sums, store ctx as bf16 (B,T,D)
  #pragma unroll
  for (int r = 0; r < 8; ++r) {
    const float inv = 1.0f / rsum[r];
    const int row = qrow_lo + r + hi * 8;
    const size_t base = ((size_t)b * TQ + row) * DQ + h * DHQ;
    #pragma unroll
    for (int dt = 0; dt < 4; ++dt)
      ctx[base + dt * 16 + (lane & 15)] = f2bf(o[dt][r] * inv);
  }
}

// ---------------------------------------------------------------------------
// Host-side launcher
// ---------------------------------------------------------------------------
extern "C" void kernel_launch(void* const* d_in, const int* in_sizes, int n_in,
                              void* d_out, int out_size, void* d_ws, size_t ws_size,
                              hipStream_t stream) {
  const float* x  = (const float*)d_in[0];
  const float* g  = (const float*)d_in[1];
  const float* be = (const float*)d_in[2];
  const float* Wq = (const float*)d_in[3];
  const float* Wo = (const float*)d_in[4];
  float* out = (float*)d_out;

  char* ws = (char*)d_ws;
  unsigned short* Wt_qkv = (unsigned short*)ws;  ws += (size_t)1024 * 3072 * 2;  //  6 MB
  unsigned short* Wt_o   = (unsigned short*)ws;  ws += (size_t)1024 * 1024 * 2;  //  2 MB
  unsigned short* xn     = (unsigned short*)ws;  ws += (size_t)8192 * 1024 * 2;  // 16 MB
  unsigned short* qkvb   = (unsigned short*)ws;  ws += (size_t)8192 * 3072 * 2;  // 48 MB
  unsigned short* KtB    = (unsigned short*)ws;  ws += (size_t)8192 * 1024 * 2;  // 16 MB
  unsigned short* ctx    = xn;                   // alias: xn dead after QKV GEMM

  wt_kernel<<<(3072 / 32) * (1024 / 32), 256, 0, stream>>>(Wq, Wt_qkv, 1024, 3072);
  wt_kernel<<<(1024 / 32) * (1024 / 32), 256, 0, stream>>>(Wo, Wt_o, 1024, 1024);
  ln_kernel<<<BQ * TQ, 256, 0, stream>>>(x, g, be, xn);
  gemm_kernel<0><<<dim3(3072 / 256, (BQ * TQ) / 128), 256, 0, stream>>>(
      xn, Wt_qkv, qkvb, KtB, nullptr, BQ * TQ, 3072, 1024);
  attn_kernel<<<dim3(TQ / 128, BQ * HQ), 256, 0, stream>>>(qkvb, KtB, ctx);
  gemm_kernel<1><<<dim3(1024 / 256, (BQ * TQ) / 128), 256, 0, stream>>>(
      ctx, Wt_o, nullptr, nullptr, out, BQ * TQ, 1024, 1024);
}
